// W4A8OF16LinearDynamicInputScale_66486093742537
// MI455X (gfx1250) — 8-bit matrix operand order repaired and checked by reading, compiled, not run on hardware
//
#include <hip/hip_runtime.h>
#include <cstdint>
#include <cstddef>

#define M_DIM 64
#define K_DIM 4096
#define N_DIM 14336
#define GROUP_SZ 128
#define G_DIM 32

typedef __attribute__((ext_vector_type(8))) int   v8i;
typedef __attribute__((ext_vector_type(2))) short s16x2;

__global__ __launch_bounds__(256) void pack_x_swizzle_kernel(
    const int* __restrict__ x,
    char* __restrict__ xswz)
{
    const int t = blockIdx.x * 256 + threadIdx.x;
    const int m = t & 63;
    const int h = (t >> 6) & 1;
    const int s = t >> 7;

    const int* __restrict__ xr = x + m * K_DIM + s * 64 + h * 16;

    int dw[8];
#pragma unroll
    for (int j = 0; j < 8; ++j) {
        const int koff = (j >> 2) * 32 + (j & 3) * 4;
        const int4 v = *reinterpret_cast<const int4*>(xr + koff);
        dw[j] = (v.x & 255) | ((v.y & 255) << 8) | ((v.z & 255) << 16) | (v.w << 24);
    }
    int4* dst = reinterpret_cast<int4*>(xswz + (size_t)t * 32);
    dst[0] = int4{dw[0], dw[1], dw[2], dw[3]};
    dst[1] = int4{dw[4], dw[5], dw[6], dw[7]};
}

__device__ __forceinline__ int dq4_pk(const int4 q, const s16x2 sv, const s16x2 zv) {
    const unsigned p01 = __builtin_amdgcn_perm((unsigned)q.y, (unsigned)q.x, 0x05040100u);
    const unsigned p23 = __builtin_amdgcn_perm((unsigned)q.w, (unsigned)q.z, 0x05040100u);
    const s16x2 w01 = __builtin_bit_cast(s16x2, p01) * sv + zv;
    const s16x2 w23 = __builtin_bit_cast(s16x2, p23) * sv + zv;
    return (int)__builtin_amdgcn_perm(__builtin_bit_cast(unsigned, w23),
                                      __builtin_bit_cast(unsigned, w01),
                                      0x06040200u);
}

__global__ __launch_bounds__(128) void w4a8_gemm_kernel(
    const char* __restrict__ xswz,
    const int*  __restrict__ qweight,
    const int*  __restrict__ s2_scales,
    const int*  __restrict__ s2_zeros,
    const float* __restrict__ input_scales,
    const float* __restrict__ s1_scales,
    const float* __restrict__ bias,
    float* __restrict__ out)
{
    const int wave = threadIdx.x >> 5;
    const int lane = threadIdx.x & 31;
    const int half = lane >> 4;
    const int l16  = lane & 15;

    const int n = blockIdx.x * 64 + wave * 16 + l16;

    const int*  __restrict__ qcol = qweight + (size_t)n * K_DIM + half * 16;
    const char* __restrict__ abase = xswz + ((size_t)half * 64 + l16) * 32;

    v8i c0 = {}, c1 = {}, c2 = {}, c3 = {};

#define LOAD_A(af, base, t)                                                    \
    do {                                                                       \
        const int4 lo = *reinterpret_cast<const int4*>((base) + (t) * 512);    \
        const int4 hi = *reinterpret_cast<const int4*>((base) + (t) * 512 + 16);\
        af[0] = lo.x; af[1] = lo.y; af[2] = lo.z; af[3] = lo.w;                \
        af[4] = hi.x; af[5] = hi.y; af[6] = hi.z; af[7] = hi.w;                \
    } while (0)

    v8i a0f, a1f, a2f, a3f;
    LOAD_A(a0f, abase, 0);
    LOAD_A(a1f, abase, 1);
    LOAD_A(a2f, abase, 2);
    LOAD_A(a3f, abase, 3);

    v8i bA = {}, bB = {};

#define KSTEP(kb, bcur, bprev)                                                 \
    do {                                                                       \
         \
        const int kbn = ((kb) + 64) & (K_DIM - 1);                             \
        const char* apn = abase + (size_t)kbn * 64;                            \
        v8i a0n, a1n, a2n, a3n;                                                \
        LOAD_A(a0n, apn, 0);                                                   \
        LOAD_A(a1n, apn, 1);                                                   \
        LOAD_A(a2n, apn, 2);                                                   \
        LOAD_A(a3n, apn, 3);                                                   \
              \
        const int* qp = qcol + (kb);                                           \
        __builtin_prefetch(qp + 1024, 0, 0);                                   \
        const int4 q0 = *reinterpret_cast<const int4*>(qp + 0);                \
        const int4 q1 = *reinterpret_cast<const int4*>(qp + 4);                \
        const int4 q2 = *reinterpret_cast<const int4*>(qp + 8);                \
        const int4 q3 = *reinterpret_cast<const int4*>(qp + 12);               \
        const int4 q4 = *reinterpret_cast<const int4*>(qp + 32);               \
        const int4 q5 = *reinterpret_cast<const int4*>(qp + 36);               \
        const int4 q6 = *reinterpret_cast<const int4*>(qp + 40);               \
        const int4 q7 = *reinterpret_cast<const int4*>(qp + 44);               \
        bcur[0] = dq4_pk(q0, sv, zv); bcur[1] = dq4_pk(q1, sv, zv);            \
        bcur[2] = dq4_pk(q2, sv, zv); bcur[3] = dq4_pk(q3, sv, zv);            \
        bcur[4] = dq4_pk(q4, sv, zv); bcur[5] = dq4_pk(q5, sv, zv);            \
        bcur[6] = dq4_pk(q6, sv, zv); bcur[7] = dq4_pk(q7, sv, zv);            \
        c0 = __builtin_amdgcn_wmma_i32_16x16x64_iu8(true, a0f, true, bcur, c0, false, false); \
        c1 = __builtin_amdgcn_wmma_i32_16x16x64_iu8(true, a1f, true, bcur, c1, false, false); \
        c2 = __builtin_amdgcn_wmma_i32_16x16x64_iu8(true, a2f, true, bcur, c2, false, false); \
        c3 = __builtin_amdgcn_wmma_i32_16x16x64_iu8(true, a3f, true, bcur, c3, false, false); \
        asm volatile("" : "+v"(bprev));         \
        a0f = a0n; a1f = a1n; a2f = a2n; a3f = a3n;                            \
    } while (0)

#pragma unroll 2
    for (int g = 0; g < G_DIM; ++g) {
        const short s = (short)s2_scales[g * N_DIM + n];
        const short z = (short)s2_zeros [g * N_DIM + n];
        const s16x2 sv = { s, s };
        const s16x2 zv = { z, z };

        KSTEP(g * GROUP_SZ,      bA, bB);
        KSTEP(g * GROUP_SZ + 64, bB, bA);
    }
#undef KSTEP
#undef LOAD_A

    const float s1 = s1_scales[n];
    const float bv = bias[n];

#define EPILOG(cacc, t)                                                        \
    do {                                                                       \
        _Pragma("unroll")                                                      \
        for (int r = 0; r < 8; ++r) {                                          \
            const int mo = (t) * 16 + r + half * 8;                            \
            out[(size_t)mo * N_DIM + n] =                                      \
                (float)cacc[r] * input_scales[mo] * s1 + bv;                   \
        }                                                                      \
    } while (0)

    EPILOG(c0, 0);
    EPILOG(c1, 1);
    EPILOG(c2, 2);
    EPILOG(c3, 3);
#undef EPILOG
}

extern "C" void kernel_launch(void* const* d_in, const int* in_sizes, int n_in,
                              void* d_out, int out_size, void* d_ws, size_t ws_size,
                              hipStream_t stream) {
    (void)in_sizes; (void)n_in; (void)out_size; (void)ws_size;

    const int*   x            = (const int*)d_in[0];
    const float* input_scales = (const float*)d_in[1];
    const int*   qweight      = (const int*)d_in[3];
    const int*   s2_scales    = (const int*)d_in[4];
    const int*   s2_zeros     = (const int*)d_in[5];
    const float* s1_scales    = (const float*)d_in[6];
    const float* bias         = (const float*)d_in[7];
    float*       out          = (float*)d_out;

    char* xswz = (char*)d_ws;

    const int chunks = (K_DIM / 64) * 2 * M_DIM;
    pack_x_swizzle_kernel<<<chunks / 256, 256, 0, stream>>>(x, xswz);

    w4a8_gemm_kernel<<<N_DIM / 64, 128, 0, stream>>>(
        xswz, qweight, s2_scales, s2_zeros, input_scales, s1_scales, bias, out);
}
